// InformerAttention_77721728188861
// MI455X (gfx1250) — compile-verified, matrix operand order checked by reading, not run on hardware
//
#include <hip/hip_runtime.h>
#include <hip/hip_bf16.h>

typedef __attribute__((ext_vector_type(16))) _Float16 v16h;
typedef __attribute__((ext_vector_type(8)))  _Float16 v8h;
typedef __attribute__((ext_vector_type(8)))  float    v8f;

#define D_MODEL 1024
#define SEQ_L   2048
#define N_BATCH 2
#define N_HEADS 16
#define D_K     64
#define N_ROWS  (N_BATCH * SEQ_L)   // 4096
#define VSTRIDE 40                  // halfs per Vt row (80B, 16B aligned, dodges bank conflicts)

static __device__ __forceinline__ v16h combine(v8h lo, v8h hi) {
    v16h r;
#pragma unroll
    for (int i = 0; i < 8; ++i) { r[i] = lo[i]; r[i + 8] = hi[i]; }
    return r;
}

static __device__ __forceinline__ v8f wmma_f16(v16h a, v16h b, v8f c) {
    return __builtin_amdgcn_wmma_f32_16x16x32_f16(false, a, false, b, (short)0, c, false, false);
}

// ---------------- prep: fp32 -> fp16 conversion ----------------
__global__ void cvt_f32_f16(const float* __restrict__ src, _Float16* __restrict__ dst, int n) {
    int i = blockIdx.x * blockDim.x + threadIdx.x;
    int stride = gridDim.x * blockDim.x;
    for (; i < n; i += stride) dst[i] = (_Float16)src[i];
}

// ---------------- LayerNorm: one row per workgroup, cast to f16 ----------------
__global__ __launch_bounds__(256) void ln_kernel(const float* __restrict__ x,
                                                 const float* __restrict__ gamma,
                                                 const float* __restrict__ beta,
                                                 _Float16* __restrict__ xn) {
    const int row = blockIdx.x;
    const float* xr = x + (size_t)row * D_MODEL;
    float s = 0.f, s2 = 0.f;
    for (int i = threadIdx.x; i < D_MODEL; i += blockDim.x) {
        float v = xr[i];
        s += v; s2 += v * v;
    }
#pragma unroll
    for (int off = 16; off >= 1; off >>= 1) {
        s  += __shfl_xor(s,  off, 32);
        s2 += __shfl_xor(s2, off, 32);
    }
    __shared__ float ws[8], ws2[8];
    int wid = threadIdx.x >> 5, lane = threadIdx.x & 31;
    if (lane == 0) { ws[wid] = s; ws2[wid] = s2; }
    __syncthreads();
    if (wid == 0) {
        s  = (lane < 8) ? ws[lane]  : 0.f;
        s2 = (lane < 8) ? ws2[lane] : 0.f;
#pragma unroll
        for (int off = 4; off >= 1; off >>= 1) {
            s  += __shfl_xor(s,  off, 32);
            s2 += __shfl_xor(s2, off, 32);
        }
        if (lane == 0) { ws[0] = s; ws2[0] = s2; }
    }
    __syncthreads();
    float mu  = ws[0] * (1.0f / D_MODEL);
    float var = ws2[0] * (1.0f / D_MODEL) - mu * mu;
    float rstd = rsqrtf(var + 1e-5f);
    for (int i = threadIdx.x; i < D_MODEL; i += blockDim.x) {
        float v = (xr[i] - mu) * rstd * gamma[i] + beta[i];
        xn[(size_t)row * D_MODEL + i] = (_Float16)v;
    }
}

// ---------------- QKV projection GEMM: xn[4096,1024] @ W^T, into [B,H,L,dk] f16 ----------------
// Block: 128 threads (4 waves). Block tile M=128, N=64.
// Wave tile 32x64: 2 A fragments x 4 B fragments = 8 WMMA accumulators.
// 12 b128 loads per 8 WMMAs per k-step (vs 10 per 4 unblocked).
__global__ __launch_bounds__(128) void qkv_gemm(const _Float16* __restrict__ xn,
                                                const _Float16* __restrict__ wq,
                                                const _Float16* __restrict__ wk,
                                                const _Float16* __restrict__ wv,
                                                const float* __restrict__ bq,
                                                const float* __restrict__ bk,
                                                const float* __restrict__ bv,
                                                _Float16* __restrict__ q16,
                                                _Float16* __restrict__ k16,
                                                _Float16* __restrict__ v16_) {
    const _Float16* W   = (blockIdx.z == 0) ? wq : (blockIdx.z == 1) ? wk : wv;
    const float*   bias = (blockIdx.z == 0) ? bq : (blockIdx.z == 1) ? bk : bv;
    _Float16*      dst  = (blockIdx.z == 0) ? q16 : (blockIdx.z == 1) ? k16 : v16_;

    const int lane = threadIdx.x & 31;
    const int w    = threadIdx.x >> 5;
    const int rowBase = blockIdx.x * 128 + w * 32;
    const int nBase   = blockIdx.y * 64;
    const int l16 = lane & 15;
    const int mh  = lane >> 4;

    v8f acc[2][4] = {};
    const _Float16* arow0 = xn + (size_t)(rowBase + l16) * D_MODEL;
    const _Float16* arow1 = xn + (size_t)(rowBase + 16 + l16) * D_MODEL;

    for (int kt = 0; kt < D_MODEL / 32; ++kt) {
        const int kb = kt * 32 + mh * 8;
        __builtin_prefetch(arow0 + kb + 256, 0, 1);
        v16h a0 = combine(*(const v8h*)(arow0 + kb), *(const v8h*)(arow0 + kb + 16));
        v16h a1 = combine(*(const v8h*)(arow1 + kb), *(const v8h*)(arow1 + kb + 16));
#pragma unroll
        for (int nt = 0; nt < 4; ++nt) {
            const _Float16* brow = W + (size_t)(nBase + nt * 16 + l16) * D_MODEL;
            v16h b = combine(*(const v8h*)(brow + kb), *(const v8h*)(brow + kb + 16));
            acc[0][nt] = wmma_f16(a0, b, acc[0][nt]);
            acc[1][nt] = wmma_f16(a1, b, acc[1][nt]);
        }
    }
    // epilogue: bias add, cast f16, scatter to [B,H,L,dk]
#pragma unroll
    for (int at = 0; at < 2; ++at) {
#pragma unroll
        for (int nt = 0; nt < 4; ++nt) {
            int col = nBase + nt * 16 + l16;
            float bval = bias[col];
            int h = col >> 6;         // /64
            int d = col & 63;
#pragma unroll
            for (int r = 0; r < 8; ++r) {
                int grow = rowBase + at * 16 + r + 8 * mh;
                int bidx = grow >> 11;          // / SEQ_L
                int lpos = grow & (SEQ_L - 1);
                dst[(((size_t)(bidx * N_HEADS + h) * SEQ_L + lpos) * D_K) + d] =
                    (_Float16)(acc[at][nt][r] + bval);
            }
        }
    }
}

// ---------------- flash attention: 64 query rows / workgroup, key tiles of 32 ----------------
__global__ __launch_bounds__(128) void attn_kernel(const _Float16* __restrict__ q16,
                                                   const _Float16* __restrict__ k16,
                                                   const _Float16* __restrict__ v16_,
                                                   _Float16* __restrict__ ctx) {
    __shared__ _Float16 smem[64 * VSTRIDE + 4 * 16 * 32];
    _Float16* Vt = smem;                                   // [64 dims][VSTRIDE keys]
    const int w    = threadIdx.x >> 5;
    const int lane = threadIdx.x & 31;
    _Float16* Pl = smem + 64 * VSTRIDE + w * 16 * 32;      // per-wave P tile [16][32]

    const int bh = blockIdx.y;
    const _Float16* Qh = q16 + (size_t)bh * SEQ_L * D_K;
    const _Float16* Kh = k16 + (size_t)bh * SEQ_L * D_K;
    const _Float16* Vh = v16_ + (size_t)bh * SEQ_L * D_K;

    const int qBase = blockIdx.x * 64 + w * 16;
    const int l16 = lane & 15;
    const int mh  = lane >> 4;

    // Q fragments for k-steps 0..31 and 32..63 (reused across all key tiles)
    v16h aq[2];
#pragma unroll
    for (int s = 0; s < 2; ++s) {
        int kb = s * 32 + mh * 8;
        const _Float16* qr = Qh + (size_t)(qBase + l16) * D_K;
        aq[s] = combine(*(const v8h*)(qr + kb), *(const v8h*)(qr + kb + 16));
    }

    float m_row[8], l_row[8];
#pragma unroll
    for (int r = 0; r < 8; ++r) { m_row[r] = -1e30f; l_row[r] = 0.f; }
    v8f acc[4] = {};

    for (int kt = 0; kt < SEQ_L / 32; ++kt) {
        __syncthreads();
        {   // stage V tile transposed: Vt[d][key], 32 keys x 64 dims
            int kk   = threadIdx.x >> 2;
            int dseg = (threadIdx.x & 3) * 16;
            const _Float16* vr = Vh + (size_t)(kt * 32 + kk) * D_K + dseg;
            v8h v0 = *(const v8h*)vr;
            v8h v1 = *(const v8h*)(vr + 8);
#pragma unroll
            for (int j = 0; j < 8; ++j) {
                Vt[(dseg + j) * VSTRIDE + kk]     = v0[j];
                Vt[(dseg + 8 + j) * VSTRIDE + kk] = v1[j];
            }
        }
        __syncthreads();

        // S = Q K^T over 32 keys: two 16-key column tiles, two k-steps each
        v8f s0 = {}, s1 = {};
#pragma unroll
        for (int s = 0; s < 2; ++s) {
            int kb = s * 32 + mh * 8;
            {
                const _Float16* kr = Kh + (size_t)(kt * 32 + l16) * D_K;
                v16h b = combine(*(const v8h*)(kr + kb), *(const v8h*)(kr + kb + 16));
                s0 = wmma_f16(aq[s], b, s0);
            }
            {
                const _Float16* kr = Kh + (size_t)(kt * 32 + 16 + l16) * D_K;
                v16h b = combine(*(const v8h*)(kr + kb), *(const v8h*)(kr + kb + 16));
                s1 = wmma_f16(aq[s], b, s1);
            }
        }

        // online softmax over 32 keys; row reductions across half-wave (16 lanes)
#pragma unroll
        for (int r = 0; r < 8; ++r) {
            float x0 = s0[r] * 0.125f;   // 1/sqrt(64)
            float x1 = s1[r] * 0.125f;
            float red = fmaxf(x0, x1);
#pragma unroll
            for (int off = 8; off >= 1; off >>= 1)
                red = fmaxf(red, __shfl_xor(red, off, 32));
            float mn = fmaxf(m_row[r], red);
            float sc = __expf(m_row[r] - mn);
            float p0 = __expf(x0 - mn);
            float p1 = __expf(x1 - mn);
            float ps = p0 + p1;
#pragma unroll
            for (int off = 8; off >= 1; off >>= 1)
                ps += __shfl_xor(ps, off, 32);
            l_row[r] = l_row[r] * sc + ps;
            m_row[r] = mn;
#pragma unroll
            for (int dt = 0; dt < 4; ++dt) acc[dt][r] *= sc;
            // store P in C-layout -> LDS (row-major 16x32) for A-fragment reload
            int prow = r + 8 * mh;
            Pl[prow * 32 + l16]      = (_Float16)p0;
            Pl[prow * 32 + 16 + l16] = (_Float16)p1;
        }
        asm volatile("s_wait_dscnt 0" ::: "memory");

        // PV: P (16x32) x V (32x64) via 4 WMMAs, B fragments from transposed Vt in LDS
        {
            int kb = mh * 8;
            const _Float16* pr = Pl + l16 * 32;
            v16h pa = combine(*(const v8h*)(pr + kb), *(const v8h*)(pr + kb + 16));
#pragma unroll
            for (int dt = 0; dt < 4; ++dt) {
                const _Float16* vtr = Vt + (size_t)(dt * 16 + l16) * VSTRIDE;
                v16h b = combine(*(const v8h*)(vtr + kb), *(const v8h*)(vtr + kb + 16));
                acc[dt] = wmma_f16(pa, b, acc[dt]);
            }
        }
    }

    // epilogue: divide by softmax denominator, write ctx f16 [B*L, D]
    int b = bh / N_HEADS, h = bh % N_HEADS;
#pragma unroll
    for (int r = 0; r < 8; ++r) {
        float inv = 1.0f / l_row[r];
        int lpos = qBase + r + 8 * mh;
        size_t grow = (size_t)b * SEQ_L + lpos;
#pragma unroll
        for (int dt = 0; dt < 4; ++dt) {
            int col = h * D_K + dt * 16 + l16;
            ctx[grow * D_MODEL + col] = (_Float16)(acc[dt][r] * inv);
        }
    }
}

// ---------------- output projection + bias + residual (f32 out) ----------------
// Same 32x64 wave tile / 128x64 block tile as qkv_gemm.
__global__ __launch_bounds__(128) void oproj_gemm(const _Float16* __restrict__ ctx,
                                                  const _Float16* __restrict__ wo,
                                                  const float* __restrict__ bo,
                                                  const float* __restrict__ x,
                                                  float* __restrict__ out) {
    const int lane = threadIdx.x & 31;
    const int w    = threadIdx.x >> 5;
    const int rowBase = blockIdx.x * 128 + w * 32;
    const int nBase   = blockIdx.y * 64;
    const int l16 = lane & 15;
    const int mh  = lane >> 4;

    v8f acc[2][4] = {};
    const _Float16* arow0 = ctx + (size_t)(rowBase + l16) * D_MODEL;
    const _Float16* arow1 = ctx + (size_t)(rowBase + 16 + l16) * D_MODEL;

    for (int kt = 0; kt < D_MODEL / 32; ++kt) {
        const int kb = kt * 32 + mh * 8;
        __builtin_prefetch(arow0 + kb + 256, 0, 1);
        v16h a0 = combine(*(const v8h*)(arow0 + kb), *(const v8h*)(arow0 + kb + 16));
        v16h a1 = combine(*(const v8h*)(arow1 + kb), *(const v8h*)(arow1 + kb + 16));
#pragma unroll
        for (int nt = 0; nt < 4; ++nt) {
            const _Float16* brow = wo + (size_t)(nBase + nt * 16 + l16) * D_MODEL;
            v16h b = combine(*(const v8h*)(brow + kb), *(const v8h*)(brow + kb + 16));
            acc[0][nt] = wmma_f16(a0, b, acc[0][nt]);
            acc[1][nt] = wmma_f16(a1, b, acc[1][nt]);
        }
    }
#pragma unroll
    for (int at = 0; at < 2; ++at) {
#pragma unroll
        for (int nt = 0; nt < 4; ++nt) {
            int col = nBase + nt * 16 + l16;
            float bval = bo[col];
#pragma unroll
            for (int r = 0; r < 8; ++r) {
                size_t grow = (size_t)(rowBase + at * 16 + r + 8 * mh);
                out[grow * D_MODEL + col] = x[grow * D_MODEL + col] + bval + acc[at][nt][r];
            }
        }
    }
}

extern "C" void kernel_launch(void* const* d_in, const int* in_sizes, int n_in,
                              void* d_out, int out_size, void* d_ws, size_t ws_size,
                              hipStream_t stream) {
    const float* x     = (const float*)d_in[0];
    const float* Wq    = (const float*)d_in[1];
    const float* bq    = (const float*)d_in[2];
    const float* Wk    = (const float*)d_in[3];
    const float* bk    = (const float*)d_in[4];
    const float* Wv    = (const float*)d_in[5];
    const float* bv    = (const float*)d_in[6];
    const float* Wo    = (const float*)d_in[7];
    const float* bo    = (const float*)d_in[8];
    const float* gamma = (const float*)d_in[9];
    const float* beta  = (const float*)d_in[10];
    float* out = (float*)d_out;

    const size_t MB = 1024ull * 1024ull;
    char* ws = (char*)d_ws;
    _Float16* wq16  = (_Float16*)(ws + 0 * MB);   // 2 MB each
    _Float16* wk16  = (_Float16*)(ws + 2 * MB);
    _Float16* wv16  = (_Float16*)(ws + 4 * MB);
    _Float16* wo16  = (_Float16*)(ws + 6 * MB);
    _Float16* xn16  = (_Float16*)(ws + 8 * MB);   // 8 MB
    _Float16* q16   = (_Float16*)(ws + 16 * MB);  // 8 MB
    _Float16* k16   = (_Float16*)(ws + 24 * MB);  // 8 MB
    _Float16* v16_  = (_Float16*)(ws + 32 * MB);  // 8 MB
    _Float16* ctx16 = (_Float16*)(ws + 40 * MB);  // 8 MB

    // 1) weight fp32->fp16
    cvt_f32_f16<<<512, 256, 0, stream>>>(Wq, wq16, D_MODEL * D_MODEL);
    cvt_f32_f16<<<512, 256, 0, stream>>>(Wk, wk16, D_MODEL * D_MODEL);
    cvt_f32_f16<<<512, 256, 0, stream>>>(Wv, wv16, D_MODEL * D_MODEL);
    cvt_f32_f16<<<512, 256, 0, stream>>>(Wo, wo16, D_MODEL * D_MODEL);

    // 2) pre-LayerNorm + cast
    ln_kernel<<<N_ROWS, 256, 0, stream>>>(x, gamma, beta, xn16);

    // 3) QKV projections (WMMA)
    qkv_gemm<<<dim3(N_ROWS / 128, D_MODEL / 64, 3), 128, 0, stream>>>(
        xn16, wq16, wk16, wv16, bq, bk, bv, q16, k16, v16_);

    // 4) flash attention (WMMA)
    attn_kernel<<<dim3(SEQ_L / 64, N_BATCH * N_HEADS), 128, 0, stream>>>(
        q16, k16, v16_, ctx16);

    // 5) output projection + bias + residual (WMMA)
    oproj_gemm<<<dim3(N_ROWS / 128, D_MODEL / 64), 128, 0, stream>>>(
        ctx16, wo16, bo, x, out);
}
